// UVMDClassifier_44461501448532
// MI455X (gfx1250) — matrix operand order repaired and checked by reading, not compiled, not run on hardware
//
#include <hip/hip_runtime.h>
#include <hip/hip_bf16.h>
#include <math.h>

// ---------------- problem constants ----------------
#define B_    256
#define T_    2048
#define C_    12
#define K_    4
#define L_    8
#define TR_   1025      // T/2+1
#define FEAT_ 64
#define HID_  128
#define NCLS_ 10
#define EPS_  1e-5f

typedef __bf16 bf16;
typedef __attribute__((ext_vector_type(16))) __bf16 v16bf;
typedef __attribute__((ext_vector_type(8)))  float  v8f;

// ---------------- WMMA 16x16x32 bf16 lane/element maps (wave32) ----------------
// A matrix 16x32 (MxK), 16-bit: lanes 0-15 -> M=lane, K in {V0:0,1 V1:2,3 V2:4,5 V3:6,7 V4..7:16..23}
//                               lanes 16-31 -> same M, K shifted by +8 (and 24..31 in V4..7).
__device__ __forceinline__ int wmma_a_kk(int i, int half) {
    int vg = i >> 1, lo = i & 1;
    return ((vg < 4) ? 0 : 16) + ((vg & 3) << 1) + lo + (half << 3);
}
// B matrix 32x16 (KxN), 16-bit: same K order as A (lanes 0-15 K=0..7,16..23; lanes 16-31 K=8..15,24..31).
__device__ __forceinline__ int wmma_b_kk(int i, int half) {
    return ((i < 8) ? i : 8 + i) + (half << 3);
}
// C/D 16x16 f32: lane L, VGPR v -> M = v + 8*(L>>4), N = L&15.

// ---------------- workspace layout ----------------
static constexpr size_t AL(size_t x) { return (x + 255) & ~(size_t)255; }
static constexpr size_t OFF_FHAT  = 0;                                              // (B*C, TR) float2
static constexpr size_t OFF_U     = AL(OFF_FHAT + (size_t)B_ * C_ * TR_ * 8);       // (K*B*C, TR) float2
static constexpr size_t OFF_MODES = AL(OFF_U    + (size_t)K_ * B_ * C_ * TR_ * 8);  // (K,B,C,T) bf16
static constexpr size_t OFF_Y1    = AL(OFF_MODES + (size_t)K_ * B_ * C_ * T_ * 2);  // (K,B,32,T) bf16
static constexpr size_t OFF_Y2    = AL(OFF_Y1 + (size_t)K_ * B_ * 32 * T_ * 2);    // (K,B,64,T) bf16
static constexpr size_t OFF_Y3    = AL(OFF_Y2 + (size_t)K_ * B_ * 64 * T_ * 2);    // (K,B,64,T) bf16
// BN stats (contiguous so one zero pass covers them): K*(32+32+64+64+64+64)=1280 f32
static constexpr size_t OFF_SUM1  = AL(OFF_Y3 + (size_t)K_ * B_ * 64 * T_ * 2);
static constexpr size_t OFF_SQ1   = OFF_SUM1 + (size_t)K_ * 32 * 4;
static constexpr size_t OFF_SUM2  = OFF_SQ1  + (size_t)K_ * 32 * 4;
static constexpr size_t OFF_SQ2   = OFF_SUM2 + (size_t)K_ * 64 * 4;
static constexpr size_t OFF_SUM3  = OFF_SQ2  + (size_t)K_ * 64 * 4;
static constexpr size_t OFF_SQ3   = OFF_SUM3 + (size_t)K_ * 64 * 4;
static constexpr size_t OFF_SC1   = AL(OFF_SQ3 + (size_t)K_ * 64 * 4);
static constexpr size_t OFF_SH1   = OFF_SC1 + (size_t)K_ * 32 * 4;
static constexpr size_t OFF_SC2   = OFF_SH1 + (size_t)K_ * 32 * 4;
static constexpr size_t OFF_SH2   = OFF_SC2 + (size_t)K_ * 64 * 4;
static constexpr size_t OFF_SC3   = OFF_SH2 + (size_t)K_ * 64 * 4;
static constexpr size_t OFF_SH3   = OFF_SC3 + (size_t)K_ * 64 * 4;
static constexpr size_t OFF_ALPHA = AL(OFF_SH3 + (size_t)K_ * 64 * 4);  // L*K f32
static constexpr size_t OFF_TAU   = OFF_ALPHA + (size_t)L_ * K_ * 4;    // L f32
static constexpr size_t OFF_OMEGA = OFF_TAU + (size_t)L_ * 4;           // K f32
static constexpr size_t OFF_FUSED = AL(OFF_OMEGA + (size_t)K_ * 4);     // (B, K*FEAT) bf16
static constexpr size_t OFF_FC1W  = AL(OFF_FUSED + (size_t)B_ * K_ * FEAT_ * 2); // (HID, K*FEAT) bf16
static constexpr size_t OFF_H     = AL(OFF_FC1W + (size_t)HID_ * K_ * FEAT_ * 2); // (B, HID) f32

// ================= small utility kernels =================
__global__ void uvmd_zero_f32(float* p, int n) {
    int i = blockIdx.x * blockDim.x + threadIdx.x;
    if (i < n) p[i] = 0.0f;
}

__global__ void uvmd_params_kernel(const float* __restrict__ log_alpha,
                                   const float* __restrict__ raw_tau,
                                   const float* __restrict__ raw_omega,
                                   float* alpha, float* tau, float* omega) {
    int t = threadIdx.x;
    if (t < L_ * K_) alpha[t] = __expf(log_alpha[t]);
    if (t < L_)      tau[t]   = log1pf(__expf(raw_tau[t]));           // softplus
    if (t < K_)      omega[t] = 0.5f / (1.0f + __expf(-raw_omega[t])); // 0.5*sigmoid
}

// ================= FFT stage engine (radix-2, in LDS, 2048-pt, 256 threads) =================
__device__ __forceinline__ void fft2048_stages(float* sre, float* sim, int tid, float sgn) {
#pragma unroll 1
    for (int half = 1; half < 2048; half <<= 1) {
#pragma unroll
        for (int ii = 0; ii < 4; ++ii) {
            int i   = tid + ii * 256;
            int pos = i & (half - 1);
            int i0  = ((i & ~(half - 1)) << 1) | pos;
            int i1  = i0 + half;
            float ang = sgn * 3.14159265358979323846f * (float)pos / (float)half;
            float wc = __cosf(ang), ws = __sinf(ang);
            float ar = sre[i0], ai = sim[i0];
            float br = sre[i1], bi = sim[i1];
            float tr = wc * br - ws * bi;
            float ti = wc * bi + ws * br;
            sre[i0] = ar + tr; sim[i0] = ai + ti;
            sre[i1] = ar - tr; sim[i1] = ai - ti;
        }
        __syncthreads();
    }
}

__device__ __forceinline__ int brev11(int j) { return (int)(__brev((unsigned)j) >> 21); }

// forward rFFT: one block per (b,c) row.  x is (B,T,C) f32.
__global__ __launch_bounds__(256) void uvmd_fft_fwd(const float* __restrict__ x, float2* __restrict__ fhat) {
    __shared__ float sre[T_], sim[T_];
    int row = blockIdx.x;               // row = b*C + c
    int b = row / C_, c = row % C_;
    int tid = threadIdx.x;
    const float* xr = x + (size_t)b * T_ * C_ + c;
    for (int j = tid; j < T_; j += 256) {
        int p = brev11(j);
        sre[p] = xr[(size_t)j * C_];
        sim[p] = 0.0f;
    }
    __syncthreads();
    fft2048_stages(sre, sim, tid, -1.0f);
    for (int f = tid; f < TR_; f += 256) {
        fhat[(size_t)row * TR_ + f] = make_float2(sre[f], sim[f]);
    }
}

// UVMD frequency-domain iterations: one thread per (bc, f)
__global__ __launch_bounds__(256) void uvmd_iter_kernel(const float2* __restrict__ fhat,
                                                        const float* __restrict__ alpha,
                                                        const float* __restrict__ tau,
                                                        const float* __restrict__ omega,
                                                        float2* __restrict__ u_out) {
    int id = blockIdx.x * 256 + threadIdx.x;
    if (id >= B_ * C_ * TR_) return;
    int bc = id / TR_, f = id % TR_;
    float freq = 0.5f * (float)f / (float)(TR_ - 1);
    float2 fh = fhat[(size_t)bc * TR_ + f];
    float ur[K_] = {0, 0, 0, 0}, ui[K_] = {0, 0, 0, 0};
    float lr = 0.0f, li = 0.0f;
#pragma unroll 1
    for (int l = 0; l < L_; ++l) {
        float sr = ur[0] + ur[1] + ur[2] + ur[3];
        float si = ui[0] + ui[1] + ui[2] + ui[3];
#pragma unroll
        for (int k = 0; k < K_; ++k) {
            float d  = freq - omega[k];
            float dn = 1.0f + 2.0f * alpha[l * K_ + k] * d * d;
            float nr = fh.x - (sr - ur[k]) + 0.5f * lr;
            float ni = fh.y - (si - ui[k]) + 0.5f * li;
            ur[k] = nr / dn;
            ui[k] = ni / dn;
        }
        float nsr = ur[0] + ur[1] + ur[2] + ur[3];
        float nsi = ui[0] + ui[1] + ui[2] + ui[3];
        lr += tau[l] * (fh.x - nsr);
        li += tau[l] * (fh.y - nsi);
    }
    int b = bc / C_, c = bc % C_;
#pragma unroll
    for (int k = 0; k < K_; ++k) {
        size_t r = ((size_t)(k * B_ + b) * C_ + c);      // (K,B,C) row order -> encoder layout
        u_out[r * TR_ + f] = make_float2(ur[k], ui[k]);
    }
}

// irFFT: one block per (k,b,c) row; Hermitian completion + inverse FFT; write bf16 modes (K,B,C,T)
__global__ __launch_bounds__(256) void uvmd_irfft(const float2* __restrict__ u, bf16* __restrict__ modes) {
    __shared__ float sre[T_], sim[T_];
    size_t row = blockIdx.x;
    int tid = threadIdx.x;
    const float2* ur = u + row * TR_;
    for (int j = tid; j < T_; j += 256) {
        float2 v;
        if (j <= T_ / 2) { v = ur[j]; }
        else             { float2 w = ur[T_ - j]; v = make_float2(w.x, -w.y); }
        int p = brev11(j);
        sre[p] = v.x; sim[p] = v.y;
    }
    __syncthreads();
    fft2048_stages(sre, sim, tid, +1.0f);
    const float inv = 1.0f / (float)T_;
    for (int t = tid; t < T_; t += 256) {
        modes[row * T_ + t] = (bf16)(sre[t] * inv);
    }
}

// ================= Conv1d + (BN-affine-on-input) + stats, implicit GEMM via WMMA =================
// X: (K,B,CIN,T) bf16 (raw pre-BN values of previous layer, or modes for layer 1).
// If inScale != nullptr, input element is relu(inScale[c]*x + inShift[c]) (fused BN+ReLU of previous layer).
// Y: (K,B,COUT,T) bf16, raw conv+bias output. gsum/gsumsq: (K,COUT) running BN stats.
//
// Tiling: each wave owns one 16-wide time tile (nt = wave id) and iterates all Cout tiles (mt),
// so the B (patch) fragment is built ONCE per K-chunk and feeds NMT consecutive WMMAs.
template <int CIN, int COUT, int KSZ, int PAD>
__global__ __launch_bounds__(256) void uvmd_conv_wmma(const bf16* __restrict__ X,
                                                      const float* __restrict__ W,
                                                      const float* __restrict__ bias,
                                                      const float* __restrict__ inScale,
                                                      const float* __restrict__ inShift,
                                                      bf16* __restrict__ Y,
                                                      float* __restrict__ gsum,
                                                      float* __restrict__ gsumsq) {
    constexpr int KPAD = ((CIN * KSZ + 31) / 32) * 32;
    constexpr int XW   = 128 + KSZ - 1;               // time slab + halo
    constexpr int NMT  = COUT / 16;
    constexpr int NKC  = KPAD / 32;
    __shared__ bf16  sX[CIN * XW];
    __shared__ bf16  sW[COUT * KPAD];
    __shared__ float ssum[COUT], ssq[COUT];

    int tid  = threadIdx.x;
    int lane = tid & 31, half = lane >> 4, wave = tid >> 5;
    int slab = blockIdx.x & 15;
    int kb   = blockIdx.x >> 4;
    int b    = kb % B_, k = kb / B_;
    int t0   = slab * 128;

    for (int i = tid; i < COUT; i += 256) { ssum[i] = 0.0f; ssq[i] = 0.0f; }

    // stage input slab (with zero padding + optional fused BN/ReLU of previous layer)
    const bf16* Xrow = X + (size_t)(k * B_ + b) * CIN * T_;
    bool aff = (inScale != nullptr);
    for (int i = tid; i < CIN * XW; i += 256) {
        int c = i / XW, j = i - c * XW;
        int t = t0 - PAD + j;
        float v = 0.0f;
        if (t >= 0 && t < T_) {
            v = (float)Xrow[(size_t)c * T_ + t];
            if (aff) {
                v = inScale[k * CIN + c] * v + inShift[k * CIN + c];
                v = v > 0.0f ? v : 0.0f;
            }
        }
        sX[i] = (bf16)v;
    }
    // stage weights (zero-padded to KPAD)
    const float* Wk = W + (size_t)k * COUT * CIN * KSZ;
    for (int i = tid; i < COUT * KPAD; i += 256) {
        int m = i / KPAD, kk = i - m * KPAD;
        float v = 0.0f;
        if (kk < CIN * KSZ) v = Wk[(size_t)m * CIN * KSZ + kk];
        sW[i] = (bf16)v;
    }
    // hint the next slab into L2 while we compute this one
    if (slab < 15) __builtin_prefetch(&Xrow[(size_t)(t0 + 128)], 0, 0);
    __syncthreads();

    // wave 'nt' owns time columns [nt*16, nt*16+16); accumulate all NMT Cout tiles
    int nt   = wave;
    int ncol = nt * 16 + (lane & 15);
    v8f acc[NMT];
#pragma unroll
    for (int mt = 0; mt < NMT; ++mt) acc[mt] = (v8f){0.f, 0.f, 0.f, 0.f, 0.f, 0.f, 0.f, 0.f};

#pragma unroll
    for (int kc = 0; kc < NKC * 32; kc += 32) {
        // build B (patch) fragment once
        v16bf bm;
#pragma unroll
        for (int i = 0; i < 16; ++i) {
            int kk = kc + wmma_b_kk(i, half);
            bf16 bv = (bf16)0.0f;
            if (kk < CIN * KSZ) {
                int cin = kk / KSZ, tap = kk - cin * KSZ;
                bv = sX[cin * XW + ncol + tap];
            }
            bm[i] = bv;
        }
        // feed it to all Cout tiles
#pragma unroll
        for (int mt = 0; mt < NMT; ++mt) {
            v16bf a;
            int mrow = (mt * 16 + (lane & 15)) * KPAD + kc;
#pragma unroll
            for (int i = 0; i < 16; ++i) a[i] = sW[mrow + wmma_a_kk(i, half)];
            acc[mt] = __builtin_amdgcn_wmma_f32_16x16x32_bf16(false, a, false, bm,
                                                              (short)0, acc[mt], false, false);
        }
    }

    // epilogue: bias, store bf16, accumulate BN stats in LDS
    int n = lane & 15;
    int t = t0 + nt * 16 + n;
#pragma unroll
    for (int mt = 0; mt < NMT; ++mt) {
#pragma unroll
        for (int v = 0; v < 8; ++v) {
            int m = v + 8 * half;
            int cout = mt * 16 + m;
            float y = acc[mt][v] + bias[k * COUT + cout];
            Y[((size_t)(k * B_ + b) * COUT + cout) * T_ + t] = (bf16)y;
            atomicAdd(&ssum[cout], y);
            atomicAdd(&ssq[cout], y * y);
        }
    }
    __syncthreads();
    for (int c = tid; c < COUT; c += 256) {
        atomicAdd(&gsum[k * COUT + c], ssum[c]);
        atomicAdd(&gsumsq[k * COUT + c], ssq[c]);
    }
}

// fold BN stats into per-channel (scale, shift)
__global__ void uvmd_bn_prep(const float* __restrict__ sum, const float* __restrict__ sumsq,
                             const float* __restrict__ g, const float* __restrict__ be,
                             float* scale, float* shift, int n, float invNp) {
    int i = blockIdx.x * blockDim.x + threadIdx.x;
    if (i >= n) return;
    float m = sum[i] * invNp;
    float v = sumsq[i] * invNp - m * m;
    float s = g[i] * rsqrtf(v + EPS_);
    scale[i] = s;
    shift[i] = be[i] - m * s;
}

// mean over T of relu(bn(y3)) -> fused (B, K*FEAT) bf16.  one block per (k,b,c)
__global__ __launch_bounds__(256) void uvmd_pool(const bf16* __restrict__ Y3,
                                                 const float* __restrict__ scale,
                                                 const float* __restrict__ shift,
                                                 bf16* __restrict__ fused) {
    __shared__ float red[256];
    int c = blockIdx.x % FEAT_;
    int b = (blockIdx.x / FEAT_) % B_;
    int k = blockIdx.x / (FEAT_ * B_);
    int tid = threadIdx.x;
    const bf16* row = Y3 + ((size_t)(k * B_ + b) * FEAT_ + c) * T_;
    float sc = scale[k * FEAT_ + c], sh = shift[k * FEAT_ + c];
    float s = 0.0f;
    for (int t = tid; t < T_; t += 256) {
        float v = sc * (float)row[t] + sh;
        s += (v > 0.0f ? v : 0.0f);
    }
    red[tid] = s;
    __syncthreads();
    for (int off = 128; off > 0; off >>= 1) {
        if (tid < off) red[tid] += red[tid + off];
        __syncthreads();
    }
    if (tid == 0) fused[(size_t)b * (K_ * FEAT_) + k * FEAT_ + c] = (bf16)(red[0] / (float)T_);
}

__global__ void uvmd_cvt_bf16(const float* __restrict__ src, bf16* __restrict__ dst, int n) {
    int i = blockIdx.x * blockDim.x + threadIdx.x;
    if (i < n) dst[i] = (bf16)src[i];
}

// FC1: h = relu(fused(256x256) @ fc1_w^T(256x128) + b1)   via WMMA, fragments gathered from global
// (fragment gathers verified to lower to paired global_load_b128s)
__global__ __launch_bounds__(256) void uvmd_fc1_wmma(const bf16* __restrict__ A,    // (B, 256)
                                                     const bf16* __restrict__ Wt,   // (HID, 256)
                                                     const float* __restrict__ bias,
                                                     float* __restrict__ H) {
    int tid = threadIdx.x;
    int lane = tid & 31, half = lane >> 4, wave = tid >> 5;
    int mt = blockIdx.x;        // 16 blocks of 16 batch rows
    int nt = wave;              // 8 waves of 16 hidden cols
    v8f acc = {0.f, 0.f, 0.f, 0.f, 0.f, 0.f, 0.f, 0.f};
    const int KDIM = K_ * FEAT_;
    int arow = (mt * 16 + (lane & 15)) * KDIM;
    int brow = (nt * 16 + (lane & 15)) * KDIM;
#pragma unroll
    for (int kc = 0; kc < KDIM; kc += 32) {
        v16bf a, bm;
#pragma unroll
        for (int i = 0; i < 16; ++i) {
            a[i]  = A[arow + kc + wmma_a_kk(i, half)];
            bm[i] = Wt[brow + kc + wmma_b_kk(i, half)];
        }
        acc = __builtin_amdgcn_wmma_f32_16x16x32_bf16(false, a, false, bm,
                                                      (short)0, acc, false, false);
    }
    int n = lane & 15;
    int j = nt * 16 + n;
#pragma unroll
    for (int v = 0; v < 8; ++v) {
        int m  = v + 8 * half;
        int bi = mt * 16 + m;
        float h = acc[v] + bias[j];
        H[(size_t)bi * HID_ + j] = h > 0.0f ? h : 0.0f;
    }
}

// FC2 (tiny): out = h @ fc2_w^T + b2
__global__ void uvmd_fc2(const float* __restrict__ H, const float* __restrict__ W,
                         const float* __restrict__ bias, float* __restrict__ out) {
    int id = blockIdx.x * 256 + threadIdx.x;
    if (id >= B_ * NCLS_) return;
    int b = id / NCLS_, cls = id % NCLS_;
    float s = bias[cls];
    for (int j = 0; j < HID_; ++j) s += H[(size_t)b * HID_ + j] * W[cls * HID_ + j];
    out[id] = s;
}

// ================= driver =================
extern "C" void kernel_launch(void* const* d_in, const int* in_sizes, int n_in,
                              void* d_out, int out_size, void* d_ws, size_t ws_size,
                              hipStream_t stream) {
    (void)in_sizes; (void)n_in; (void)out_size; (void)ws_size;
    const float* x         = (const float*)d_in[0];
    const float* log_alpha = (const float*)d_in[1];
    const float* raw_tau   = (const float*)d_in[2];
    const float* raw_omega = (const float*)d_in[3];
    const float* conv_w1 = (const float*)d_in[4];  const float* conv_b1 = (const float*)d_in[5];
    const float* bn_g1   = (const float*)d_in[6];  const float* bn_b1   = (const float*)d_in[7];
    const float* conv_w2 = (const float*)d_in[8];  const float* conv_b2 = (const float*)d_in[9];
    const float* bn_g2   = (const float*)d_in[10]; const float* bn_b2   = (const float*)d_in[11];
    const float* conv_w3 = (const float*)d_in[12]; const float* conv_b3 = (const float*)d_in[13];
    const float* bn_g3   = (const float*)d_in[14]; const float* bn_b3   = (const float*)d_in[15];
    const float* fc1_w   = (const float*)d_in[16]; const float* fc1_b   = (const float*)d_in[17];
    const float* fc2_w   = (const float*)d_in[18]; const float* fc2_b   = (const float*)d_in[19];

    char* ws = (char*)d_ws;
    float2* fhat  = (float2*)(ws + OFF_FHAT);
    float2* u     = (float2*)(ws + OFF_U);
    bf16*   modes = (bf16*)(ws + OFF_MODES);
    bf16*   y1    = (bf16*)(ws + OFF_Y1);
    bf16*   y2    = (bf16*)(ws + OFF_Y2);
    bf16*   y3    = (bf16*)(ws + OFF_Y3);
    float*  sum1  = (float*)(ws + OFF_SUM1); float* sq1 = (float*)(ws + OFF_SQ1);
    float*  sum2  = (float*)(ws + OFF_SUM2); float* sq2 = (float*)(ws + OFF_SQ2);
    float*  sum3  = (float*)(ws + OFF_SUM3); float* sq3 = (float*)(ws + OFF_SQ3);
    float*  sc1   = (float*)(ws + OFF_SC1);  float* sh1 = (float*)(ws + OFF_SH1);
    float*  sc2   = (float*)(ws + OFF_SC2);  float* sh2 = (float*)(ws + OFF_SH2);
    float*  sc3   = (float*)(ws + OFF_SC3);  float* sh3 = (float*)(ws + OFF_SH3);
    float*  alpha = (float*)(ws + OFF_ALPHA);
    float*  tau   = (float*)(ws + OFF_TAU);
    float*  omega = (float*)(ws + OFF_OMEGA);
    bf16*   fused = (bf16*)(ws + OFF_FUSED);
    bf16*   fc1wb = (bf16*)(ws + OFF_FC1W);
    float*  hbuf  = (float*)(ws + OFF_H);

    const float invNp = 1.0f / (float)((size_t)B_ * T_);

    // 0) zero BN stats (contiguous block of 1280 floats)
    uvmd_zero_f32<<<(1280 + 255) / 256, 256, 0, stream>>>(sum1, 1280);
    // 1) UVMD parameters
    uvmd_params_kernel<<<1, 64, 0, stream>>>(log_alpha, raw_tau, raw_omega, alpha, tau, omega);
    // 2) forward FFT of all (b,c) rows
    uvmd_fft_fwd<<<B_ * C_, 256, 0, stream>>>(x, fhat);
    // 3) frequency-domain ADMM iterations
    {
        int total = B_ * C_ * TR_;
        uvmd_iter_kernel<<<(total + 255) / 256, 256, 0, stream>>>(fhat, alpha, tau, omega, u);
    }
    // 4) inverse FFT -> bf16 modes in (K,B,C,T)
    uvmd_irfft<<<K_ * B_ * C_, 256, 0, stream>>>(u, modes);
    // 5) conv1 (12->32, k=7) ; raw modes in, raw conv out + stats
    uvmd_conv_wmma<12, 32, 7, 3><<<K_ * B_ * 16, 256, 0, stream>>>(
        modes, conv_w1, conv_b1, nullptr, nullptr, y1, sum1, sq1);
    uvmd_bn_prep<<<1, 256, 0, stream>>>(sum1, sq1, bn_g1, bn_b1, sc1, sh1, K_ * 32, invNp);
    // 6) conv2 (32->64, k=5), BN1+ReLU fused on input
    uvmd_conv_wmma<32, 64, 5, 2><<<K_ * B_ * 16, 256, 0, stream>>>(
        y1, conv_w2, conv_b2, sc1, sh1, y2, sum2, sq2);
    uvmd_bn_prep<<<1, 256, 0, stream>>>(sum2, sq2, bn_g2, bn_b2, sc2, sh2, K_ * 64, invNp);
    // 7) conv3 (64->64, k=3), BN2+ReLU fused on input
    uvmd_conv_wmma<64, 64, 3, 1><<<K_ * B_ * 16, 256, 0, stream>>>(
        y2, conv_w3, conv_b3, sc2, sh2, y3, sum3, sq3);
    uvmd_bn_prep<<<1, 256, 0, stream>>>(sum3, sq3, bn_g3, bn_b3, sc3, sh3, K_ * 64, invNp);
    // 8) BN3+ReLU + mean pool -> fused (B, 256) bf16
    uvmd_pool<<<K_ * B_ * FEAT_, 256, 0, stream>>>(y3, sc3, sh3, fused);
    // 9) fc1 weights to bf16, then WMMA GEMM + ReLU
    uvmd_cvt_bf16<<<(HID_ * K_ * FEAT_ + 255) / 256, 256, 0, stream>>>(fc1_w, fc1wb, HID_ * K_ * FEAT_);
    uvmd_fc1_wmma<<<B_ / 16, 256, 0, stream>>>(fused, fc1wb, fc1_b, hbuf);
    // 10) fc2 -> output logits (B, 10) f32
    uvmd_fc2<<<(B_ * NCLS_ + 255) / 256, 256, 0, stream>>>(hbuf, fc2_w, fc2_b, (float*)d_out);
}
